// ParticleNet_87162066305549
// MI455X (gfx1250) — matrix operand order repaired and checked by reading, not compiled, not run on hardware
//
#include <hip/hip_runtime.h>

// ---------------------------------------------------------------------------
// ParticleNet forward for MI455X (gfx1250, wave32, WMMA).
// Edge-MLP GEMMs (the ~360 GFLOP hot path) run on v_wmma_f32_16x16x32_bf16.
// MP points per workgroup amortize weight (B-operand) traffic MP x.
// ---------------------------------------------------------------------------

typedef __attribute__((ext_vector_type(8)))  float  v8f;
typedef __attribute__((ext_vector_type(16))) __bf16 v16bf;

union FragBF {
    v16bf v;
    uint4 q[2];
    unsigned short s[16];
};

static __device__ __forceinline__ unsigned short f32_to_bf16(float f) {
    unsigned int u = __float_as_uint(f);
    unsigned int lsb = (u >> 16) & 1u;
    u += 0x7fffu + lsb;                 // round to nearest even
    return (unsigned short)(u >> 16);
}

// ---------------------------------------------------------------------------
// Stage 0: per-batch hits-mean recalculation + X*=mask, and n_tracks[b]
// ---------------------------------------------------------------------------
__global__ __launch_bounds__(256) void k_hits_mean(const float* __restrict__ X,
                                                   const float* __restrict__ mask,
                                                   float* __restrict__ X0,
                                                   float* __restrict__ ntrk) {
    const int b = blockIdx.x;
    const int t = threadIdx.x;
    __shared__ float r0[256], r1[256], r2[256], rc[256], rm[256];
    __shared__ float mean3[3];

    float s0 = 0.f, s1 = 0.f, s2 = 0.f, c = 0.f, ms = 0.f;
    for (int s = t; s < 1024 * 5; s += 256) {
        int n = s / 5, h = s - n * 5;
        float m = mask[(b << 10) + n];
        const float* xp = X + ((size_t)(b << 10) + n) * 30 + h * 3;
        float v0 = xp[0] * m, v1 = xp[1] * m, v2 = xp[2] * m;
        if (v0 != 0.f && v1 != 0.f && v2 != 0.f) c += 1.f;
        s0 += v0; s1 += v1; s2 += v2;
    }
    for (int n = t; n < 1024; n += 256) ms += mask[(b << 10) + n];
    r0[t] = s0; r1[t] = s1; r2[t] = s2; rc[t] = c; rm[t] = ms;
    __syncthreads();
    for (int s = 128; s > 0; s >>= 1) {
        if (t < s) {
            r0[t] += r0[t + s]; r1[t] += r1[t + s]; r2[t] += r2[t + s];
            rc[t] += rc[t + s]; rm[t] += rm[t + s];
        }
        __syncthreads();
    }
    if (t == 0) {
        float ng = rc[0]; if (ng == 0.f) ng = 1.f;
        mean3[0] = r0[0] / ng; mean3[1] = r1[0] / ng; mean3[2] = r2[0] / ng;
        ntrk[b] = rm[0];
    }
    __syncthreads();
    for (int e = t; e < 1024 * 30; e += 256) {
        int n = e / 30, f = e - n * 30;
        float m = mask[(b << 10) + n];
        float v = (f >= 25 && f < 28) ? mean3[f - 25]
                                      : X[((size_t)(b << 10) + n) * 30 + f];
        X0[((size_t)(b << 10) + n) * 30 + f] = v * m;
    }
}

// ---------------------------------------------------------------------------
// fp32 weights [fd][fin] -> bf16 padded [fd][finpad] (zero pad K)
// ---------------------------------------------------------------------------
__global__ void k_prep_w(const float* __restrict__ W, unsigned short* __restrict__ Wt,
                         int fd, int fin, int finpad) {
    int i = blockIdx.x * 256 + threadIdx.x;
    if (i >= fd * finpad) return;
    int n = i / finpad, k = i - n * finpad;
    Wt[i] = (k < fin) ? f32_to_bf16(W[n * fin + k]) : (unsigned short)0;
}

// ---------------------------------------------------------------------------
// kNN (k=16): one block per point; dist row in LDS, 16 argmin passes
// ---------------------------------------------------------------------------
__global__ __launch_bounds__(256) void k_knn(const float* __restrict__ pts, int stride,
                                             int pdim, const float* __restrict__ mask,
                                             int* __restrict__ idxout) {
    const int bn = blockIdx.x;
    const int b = bn >> 10;
    const int t = threadIdx.x;
    __shared__ float pbuf[128];
    __shared__ float dist[1024];
    __shared__ float rv[256];
    __shared__ int   ri[256];

    const float* pn = pts + (size_t)bn * stride;
    for (int d = t; d < pdim; d += 256) pbuf[d] = pn[d];
    __syncthreads();

    for (int m = t; m < 1024; m += 256) {
        const float* pm = pts + ((size_t)(b << 10) + m) * stride;
        float s = 0.f;
        for (int d = 0; d < pdim; ++d) { float df = pbuf[d] - pm[d]; s += df * df; }
        dist[m] = (mask[(b << 10) + m] > 0.f) ? s : 3.0e38f;
    }
    __syncthreads();

    for (int it = 0; it < 16; ++it) {
        float bv = 3.9e38f; int bi = t;
        for (int m = t; m < 1024; m += 256) {
            float dv = dist[m];
            if (dv < bv || (dv == bv && m < bi)) { bv = dv; bi = m; }
        }
        rv[t] = bv; ri[t] = bi;
        __syncthreads();
        for (int s = 128; s > 0; s >>= 1) {
            if (t < s) {
                if (rv[t + s] < rv[t] || (rv[t + s] == rv[t] && ri[t + s] < ri[t])) {
                    rv[t] = rv[t + s]; ri[t] = ri[t + s];
                }
            }
            __syncthreads();
        }
        if (t == 0) { idxout[bn * 16 + it] = ri[0]; dist[ri[0]] = 4.0e38f; }
        __syncthreads();
    }
}

// ---------------------------------------------------------------------------
// Fused EdgeConv stage: MP points per workgroup. Gather 16*MP edges ->
// 4x (WMMA GEMM + bias + LN + PReLU) -> masked neighbor mean.
// Each 16-edge group of one point is one WMMA M-tile; one B fragment feeds
// MP back-to-back independent WMMAs (MP x weight reuse, no RAW stalls).
// ---------------------------------------------------------------------------
struct BlkArgs {
    const unsigned short* Wt;  // bf16 [FD][K] (row = output feature)
    const float* b;
    const float* g;
    const float* be;
    const float* a;            // scalar
};
struct EcArgs {
    BlkArgs blk[4];
    const float* xin;          // [B*N][PREV] f32
    const int*   idx;          // [B*N][16]
    const float* ntrk;         // [B]
    float*       xout;         // [B*N][FD]
};

template <int FD, int K0, int PREV, int MP>
__global__ __launch_bounds__(2 * FD) void k_edgeconv(EcArgs args) {
    constexpr int T  = 2 * FD;     // FD/16 waves * 32 lanes
    constexpr int R  = 16 * MP;    // edge rows held in LDS
    constexpr int TS = T / R;      // threads per edge-row for LN
    const int bn0  = blockIdx.x * MP;
    const int b    = bn0 >> 10;
    const int t    = threadIdx.x;
    const int lane = t & 31;
    const int w    = t >> 5;

    __shared__ __align__(16) unsigned short feat[R * K0]; // bf16 A operand (K0==FD)
    __shared__ float hbuf[R * FD];
    __shared__ float xcb[MP * PREV];
    __shared__ int   nbr[R];
    __shared__ float psum[T], psq[T];
    __shared__ float rmean[R], rinv[R];

    // --- gather centers + neighbor indices ---
    for (int d = t; d < MP * PREV; d += T) {
        int p = d / PREV, dd = d - p * PREV;
        xcb[p * PREV + dd] = args.xin[(size_t)(bn0 + p) * PREV + dd];
    }
    if (t < R) nbr[t] = args.idx[(bn0 + (t >> 4)) * 16 + (t & 15)];
    __syncthreads();

    // --- build bf16 edge features [xc, xc - xn] (zero-pad K for layer 0) ---
    for (int e = t; e < R * K0; e += T) {
        int row = e / K0, d = e - row * K0;     // row = p*16 + j
        int p = row >> 4;
        float v;
        if (d < PREV) v = xcb[p * PREV + d];
        else if (d < 2 * PREV) {
            int dd = d - PREV;
            v = xcb[p * PREV + dd] -
                args.xin[((size_t)(b << 10) + nbr[row]) * PREV + dd];
        } else v = 0.f;
        feat[row * K0 + d] = f32_to_bf16(v);
    }
    __syncthreads();

    // --- 4 blocks of Linear -> LayerNorm -> PReLU, fused in LDS ---
    for (int blk = 0; blk < 4; ++blk) {
        const int K = (blk == 0) ? K0 : FD;
        const unsigned short* Wt = args.blk[blk].Wt;
        const float* bias = args.blk[blk].b;
        const float* g    = args.blk[blk].g;
        const float* be   = args.blk[blk].be;
        const float alpha = args.blk[blk].a[0];

        // A fragment (ISA 16-bit A 16x32): lane<16 -> K {0..7,16..23}, lane>=16 -> +8
        // B fragment (ISA 16-bit B 32x16): col = lane&15, lane<16 -> K {0..7,16..23}, lane>=16 -> +8
        v8f acc[MP];
#pragma unroll
        for (int p = 0; p < MP; ++p) acc[p] = {};
        const int rowA  = lane & 15;
        const int koffA = (lane < 16) ? 0 : 8;
        const int koffB = (lane < 16) ? 0 : 8;
        const int colB  = (w << 4) + (lane & 15);
        for (int kk = 0; kk < K; kk += 32) {
            FragBF Bf;
            const unsigned short* bp = Wt + (size_t)colB * K + kk + koffB;
            Bf.q[0] = *(const uint4*)(bp);       // global_load_b128
            Bf.q[1] = *(const uint4*)(bp + 16);
#pragma unroll
            for (int p = 0; p < MP; ++p) {
                FragBF A;
                const unsigned short* ap =
                    &feat[((p << 4) + rowA) * K + kk + koffA];
                A.q[0] = *(const uint4*)(ap);    // ds_load_b128
                A.q[1] = *(const uint4*)(ap + 16);
                acc[p] = __builtin_amdgcn_wmma_f32_16x16x32_bf16(
                    false, A.v, false, Bf.v, (short)0, acc[p], false, false);
            }
        }
        __syncthreads();   // all waves done reading feat

        // D layout: VGPR r, lane<16 -> row r, lane>=16 -> row r+8; col = lane&15
        {
            const int col   = (w << 4) + (lane & 15);
            const int rbase = (lane < 16) ? 0 : 8;
            float bv = bias[col];
#pragma unroll
            for (int p = 0; p < MP; ++p)
#pragma unroll
                for (int r = 0; r < 8; ++r)
                    hbuf[((p << 4) + rbase + r) * FD + col] = acc[p][r] + bv;
        }
        __syncthreads();

        // LayerNorm stats per edge-row (R rows, TS threads each)
        {
            int row = t / TS, j = t - row * TS;
            float s = 0.f, sq = 0.f;
            for (int c2 = j; c2 < FD; c2 += TS) {
                float v = hbuf[row * FD + c2];
                s += v; sq += v * v;
            }
            psum[t] = s; psq[t] = sq;
        }
        __syncthreads();
        if ((t % TS) == 0) {
            int row = t / TS;
            float s = 0.f, sq = 0.f;
            for (int j = 0; j < TS; ++j) { s += psum[row * TS + j]; sq += psq[row * TS + j]; }
            float mu  = s / (float)FD;
            float var = sq / (float)FD - mu * mu;
            rmean[row] = mu;
            rinv[row]  = rsqrtf(var + 1e-5f);
        }
        __syncthreads();
        // apply LN + PReLU; keep f32 (for aggregation) and bf16 (next A operand)
        {
            int row = t / TS, j = t - (t / TS) * TS;
            float mu = rmean[row], inv = rinv[row];
            for (int c2 = j; c2 < FD; c2 += TS) {
                float v = hbuf[row * FD + c2];
                v = g[c2] * (v - mu) * inv + be[c2];
                v = (v >= 0.f) ? v : alpha * v;
                hbuf[row * FD + c2] = v;
                feat[row * FD + c2] = f32_to_bf16(v);
            }
        }
        __syncthreads();
    }

    // --- masked neighbor mean over k=16, per point ---
    const float nt  = args.ntrk[b];
    const float nts = (nt == 0.f) ? 1.f : nt;
    for (int c2 = t; c2 < FD; c2 += T) {
#pragma unroll
        for (int p = 0; p < MP; ++p) {
            float s = 0.f;
            for (int j = 0; j < 16; ++j)
                if ((float)j < nt) s += hbuf[((p << 4) + j) * FD + c2];
            args.xout[(size_t)(bn0 + p) * FD + c2] = s / nts;
        }
    }
}

// ---------------------------------------------------------------------------
// Pooling: H = [masked_mean(X3), masked_max(X3)]  ->  (32, 512)
// ---------------------------------------------------------------------------
__global__ __launch_bounds__(256) void k_pool(const float* __restrict__ X3,
                                              const float* __restrict__ mask,
                                              const float* __restrict__ ntrk,
                                              float* __restrict__ H) {
    const int b = blockIdx.x, c = threadIdx.x;   // c in [0,256)
    const float nt = ntrk[b];
    float sum = 0.f, mx = -3.9e38f, mn = 3.9e38f;
    for (int n = 0; n < 1024; ++n) {
        float v = X3[((size_t)(b << 10) + n) * 256 + c];
        float m = mask[(b << 10) + n];
        sum += v * m;
        if (m > 0.f && v > mx) mx = v;
        if (v < mn) mn = v;
    }
    float nts = (nt == 0.f) ? 1.f : nt;
    H[b * 512 + c]       = sum / nts;
    H[b * 512 + 256 + c] = (nt > 0.f) ? mx : mn;
}

// ---------------------------------------------------------------------------
// Head: h1 = PReLU(H @ W1^T + b1); BN over batch; out = h1 @ W2^T + b2
// ---------------------------------------------------------------------------
__global__ __launch_bounds__(256) void k_pred1(const float* __restrict__ H,
                                               const float* __restrict__ W1,
                                               const float* __restrict__ b1,
                                               const float* __restrict__ a,
                                               float* __restrict__ h1) {
    int o = blockIdx.x * 256 + threadIdx.x;
    if (o >= 32 * 512) return;
    int row = o >> 9, col = o & 511;
    const float* hr = H + row * 512;
    const float* wr = W1 + col * 512;
    float s = b1[col];
    for (int k2 = 0; k2 < 512; ++k2) s += hr[k2] * wr[k2];
    float al = a[0];
    h1[o] = (s >= 0.f) ? s : al * s;
}

__global__ __launch_bounds__(256) void k_bn(float* __restrict__ h1,
                                            const float* __restrict__ g,
                                            const float* __restrict__ be) {
    int f = blockIdx.x * 256 + threadIdx.x;
    if (f >= 512) return;
    float s = 0.f, sq = 0.f;
    for (int r = 0; r < 32; ++r) { float v = h1[r * 512 + f]; s += v; sq += v * v; }
    float mu  = s / 32.f;
    float var = sq / 32.f - mu * mu;
    float inv = rsqrtf(var + 1e-5f);
    for (int r = 0; r < 32; ++r) {
        float v = h1[r * 512 + f];
        h1[r * 512 + f] = g[f] * (v - mu) * inv + be[f];
    }
}

__global__ __launch_bounds__(64) void k_pred2(const float* __restrict__ h1,
                                              const float* __restrict__ W2,
                                              const float* __restrict__ b2,
                                              float* __restrict__ out) {
    int o = threadIdx.x;
    if (o >= 64) return;
    int row = o >> 1, col = o & 1;
    float s = b2[col];
    const float* hr = h1 + row * 512;
    const float* wr = W2 + col * 512;
    for (int k2 = 0; k2 < 512; ++k2) s += hr[k2] * wr[k2];
    out[o] = s;
}

// ---------------------------------------------------------------------------
// Host launch. Input flattening (setup_inputs dict order):
//   0:X  1:mask  then edgeconvs: layer L block B -> 2 + L*20 + B*5 + {W,b,g,be,a}
//   pred: 62:W1 63:b1 64:a 65:bn_g 66:bn_b 67:W2 68:b2
// ---------------------------------------------------------------------------
extern "C" void kernel_launch(void* const* d_in, const int* in_sizes, int n_in,
                              void* d_out, int out_size, void* d_ws, size_t ws_size,
                              hipStream_t stream) {
    (void)in_sizes; (void)n_in; (void)out_size; (void)ws_size;
    const float* X    = (const float*)d_in[0];
    const float* mask = (const float*)d_in[1];
    auto IN = [&](int i) { return (const float*)d_in[i]; };

    char* ws = (char*)d_ws;
    size_t off = 0;
    auto alloc = [&](size_t bytes) -> char* {
        off = (off + 255) & ~(size_t)255;
        char* p = ws + off;
        off += bytes;
        return p;
    };
    float* X0   = (float*)alloc(32ull * 1024 * 30 * 4);
    float* X1   = (float*)alloc(32ull * 1024 * 64 * 4);
    float* X2   = (float*)alloc(32ull * 1024 * 128 * 4);
    float* X3   = (float*)alloc(32ull * 1024 * 256 * 4);
    int*   IDX  = (int*)alloc(32ull * 1024 * 16 * 4);
    float* NTRK = (float*)alloc(32 * 4);
    float* Hb   = (float*)alloc(32 * 512 * 4);
    float* H1   = (float*)alloc(32 * 512 * 4);

    const int fds[3]   = {64, 128, 256};
    const int fin0s[3] = {60, 128, 256};
    unsigned short* Wt[3][4];
    for (int L = 0; L < 3; ++L)
        for (int B2 = 0; B2 < 4; ++B2)
            Wt[L][B2] = (unsigned short*)alloc((size_t)fds[L] * fds[L] * 2);

    // Stage 0 + weight conversion (bf16, K padded to fd)
    k_hits_mean<<<32, 256, 0, stream>>>(X, mask, X0, NTRK);
    for (int L = 0; L < 3; ++L)
        for (int B2 = 0; B2 < 4; ++B2) {
            int fd = fds[L];
            int fin = (B2 == 0) ? fin0s[L] : fd;
            int tot = fd * fd;
            k_prep_w<<<(tot + 255) / 256, 256, 0, stream>>>(
                IN(2 + L * 20 + B2 * 5 + 0), Wt[L][B2], fd, fin, fd);
        }

    auto fill = [&](EcArgs& a, int L, const float* xin, float* xout) {
        for (int B2 = 0; B2 < 4; ++B2) {
            a.blk[B2].Wt = Wt[L][B2];
            a.blk[B2].b  = IN(2 + L * 20 + B2 * 5 + 1);
            a.blk[B2].g  = IN(2 + L * 20 + B2 * 5 + 2);
            a.blk[B2].be = IN(2 + L * 20 + B2 * 5 + 3);
            a.blk[B2].a  = IN(2 + L * 20 + B2 * 5 + 4);
        }
        a.xin = xin; a.idx = IDX; a.ntrk = NTRK; a.xout = xout;
    };

    // EdgeConv 0: points = X0[:, :15], features = X0 (30) -> X1 (64)   [MP=4]
    k_knn<<<32 * 1024, 256, 0, stream>>>(X0, 30, 15, mask, IDX);
    { EcArgs a; fill(a, 0, X0, X1);
      k_edgeconv<64, 64, 30, 4><<<(32 * 1024) / 4, 128, 0, stream>>>(a); }

    // EdgeConv 1: points = X1 (64) -> X2 (128)                          [MP=4]
    k_knn<<<32 * 1024, 256, 0, stream>>>(X1, 64, 64, mask, IDX);
    { EcArgs a; fill(a, 1, X1, X2);
      k_edgeconv<128, 128, 64, 4><<<(32 * 1024) / 4, 256, 0, stream>>>(a); }

    // EdgeConv 2: points = X2 (128) -> X3 (256)                         [MP=2]
    k_knn<<<32 * 1024, 256, 0, stream>>>(X2, 128, 128, mask, IDX);
    { EcArgs a; fill(a, 2, X2, X3);
      k_edgeconv<256, 256, 128, 2><<<(32 * 1024) / 2, 512, 0, stream>>>(a); }

    // Pool + head
    k_pool<<<32, 256, 0, stream>>>(X3, mask, NTRK, Hb);
    k_pred1<<<(32 * 512 + 255) / 256, 256, 0, stream>>>(Hb, IN(62), IN(63), IN(64), H1);
    k_bn<<<2, 256, 0, stream>>>(H1, IN(65), IN(66));
    k_pred2<<<1, 64, 0, stream>>>(H1, IN(67), IN(68), (float*)d_out);
}
